// LiGRU_Layer_43585328120312
// MI455X (gfx1250) — matrix operand order repaired and checked by reading, not compiled, not run on hardware
//
#include <hip/hip_runtime.h>
#include <hip/hip_bf16.h>

typedef __bf16 bf16;
typedef __attribute__((ext_vector_type(16))) __bf16 v16bf;
typedef __attribute__((ext_vector_type(8)))  __bf16 v8bf;
typedef __attribute__((ext_vector_type(8)))  float  v8f;

#define LN_EPSF 1e-5f

// Problem sizes
#define B_  16
#define T_  512
#define D_  512
#define H_  1024
#define G_  (2 * H_)      // 2048 gates
#define M_  (B_ * T_)     // 8192 tokens

// Recurrence config
#define NWG        32     // persistent workgroups
#define RT_THREADS 256    // 8 waves
#define RWAVES     8
#define KCH        (H_ / RWAVES)   // 128 K per wave
#define NCOLS      32     // h columns owned per WG
#define LROWS      64     // U rows resident in LDS (32 'a' + 32 'z')
#define LPAD       8
#define LSTR       (H_ + LPAD)     // 1032 elem row stride (2064 B, bank-conflict free)

#define WMMA_BF16(A, Bm, C) \
  __builtin_amdgcn_wmma_f32_16x16x32_bf16(false, (A), false, (Bm), (short)0, (C), false, false)

__device__ __forceinline__ v8f zero8() {
  v8f z;
#pragma unroll
  for (int i = 0; i < 8; ++i) z[i] = 0.0f;
  return z;
}

// A fragment: 16x32 bf16, row-major source (stride lda elements).
// Lane L (0-15): M=L, K in {hi*8+0..7} U {16+hi*8+0..7} with hi = L>>4.
__device__ __forceinline__ v16bf load_frag_a(const bf16* A, int lda, int m0, int k0, int lane) {
  int row = lane & 15, hi = lane >> 4;
  const bf16* p = A + (size_t)(m0 + row) * lda + k0 + hi * 8;
  v8bf a0 = *(const v8bf*)p;          // K = hi*8 + 0..7
  v8bf a1 = *(const v8bf*)(p + 16);   // K = 16 + hi*8 + 0..7
  v16bf r;
#pragma unroll
  for (int i = 0; i < 8; ++i) { r[i] = a0[i]; r[i + 8] = a1[i]; }
  return r;
}

// B fragment: 32x16 bf16 (KxN), source stored N-major [n][k] (stride ldb elements).
// Lane L: N = L&15, K in {hi*8+0..7} U {16+hi*8+0..7} with hi = L>>4.
__device__ __forceinline__ v16bf load_frag_b(const bf16* Bt, int ldb, int n0, int k0, int lane) {
  int col = lane & 15, hi = lane >> 4;
  const bf16* p = Bt + (size_t)(n0 + col) * ldb + k0 + hi * 8;
  v8bf b0 = *(const v8bf*)p;
  v8bf b1 = *(const v8bf*)(p + 16);
  v16bf r;
#pragma unroll
  for (int i = 0; i < 8; ++i) { r[i] = b0[i]; r[i + 8] = b1[i]; }
  return r;
}

// ---------------- fp32 -> bf16 hi/lo split ----------------
__global__ void split_f32_bf16(const float* __restrict__ x, bf16* __restrict__ hi,
                               bf16* __restrict__ lo, int n) {
  for (int i = blockIdx.x * blockDim.x + threadIdx.x; i < n; i += gridDim.x * blockDim.x) {
    float v = x[i];
    bf16 h = (bf16)v;
    hi[i] = h;
    lo[i] = (bf16)(v - (float)h);
  }
}

// ---------------- Phase 1: w = X @ W^T (split-bf16 WMMA, fp32 accumulate) ----------------
// Each wave computes a 32x64 tile of C[8192, 2048]. 8 waves per block.
__global__ __launch_bounds__(256) void gemm_xw(const bf16* __restrict__ Xh, const bf16* __restrict__ Xl,
                                               const bf16* __restrict__ Wh, const bf16* __restrict__ Wl,
                                               float* __restrict__ Wo) {
  int wave = (blockIdx.x << 3) + (threadIdx.x >> 5);
  int lane = threadIdx.x & 31;
  int tn = wave & 31;   // 32 N-tiles of 64
  int tm = wave >> 5;   // 256 M-tiles of 32
  int m0 = tm * 32, n0 = tn * 64;

  v8f c[2][4];
#pragma unroll
  for (int s = 0; s < 2; ++s)
#pragma unroll
    for (int j = 0; j < 4; ++j) c[s][j] = zero8();

  for (int k0 = 0; k0 < D_; k0 += 32) {
    if (k0 + 32 < D_) {
      __builtin_prefetch(Xh + (size_t)m0 * D_ + k0 + 32, 0, 1);
      __builtin_prefetch(Wh + (size_t)n0 * D_ + k0 + 32, 0, 1);
    }
    v16bf ah0 = load_frag_a(Xh, D_, m0, k0, lane);
    v16bf ah1 = load_frag_a(Xh, D_, m0 + 16, k0, lane);
    v16bf al0 = load_frag_a(Xl, D_, m0, k0, lane);
    v16bf al1 = load_frag_a(Xl, D_, m0 + 16, k0, lane);
#pragma unroll
    for (int j = 0; j < 4; ++j) {
      v16bf bh = load_frag_b(Wh, D_, n0 + j * 16, k0, lane);
      v16bf bl = load_frag_b(Wl, D_, n0 + j * 16, k0, lane);
      c[0][j] = WMMA_BF16(ah0, bh, c[0][j]);
      c[0][j] = WMMA_BF16(ah0, bl, c[0][j]);
      c[0][j] = WMMA_BF16(al0, bh, c[0][j]);
      c[1][j] = WMMA_BF16(ah1, bh, c[1][j]);
      c[1][j] = WMMA_BF16(ah1, bl, c[1][j]);
      c[1][j] = WMMA_BF16(al1, bh, c[1][j]);
    }
  }
  int hi = lane >> 4, nl = lane & 15;
#pragma unroll
  for (int s = 0; s < 2; ++s)
#pragma unroll
    for (int j = 0; j < 4; ++j)
#pragma unroll
      for (int r = 0; r < 8; ++r)
        Wo[(size_t)(m0 + s * 16 + r + hi * 8) * G_ + n0 + j * 16 + nl] = c[s][j][r];
}

// ---------------- Phase 1b: in-place LayerNorm over 2048 gates ----------------
__global__ __launch_bounds__(256) void ln_kernel(float* __restrict__ w,
                                                 const float* __restrict__ gamma,
                                                 const float* __restrict__ beta) {
  __shared__ float ssum[8], ssum2[8], smu[1], srs[1];
  int row = blockIdx.x, tid = threadIdx.x;
  float* pr = w + (size_t)row * G_;
  float v[8];
  float4 a = *(const float4*)(pr + tid * 8);
  float4 b = *(const float4*)(pr + tid * 8 + 4);
  v[0] = a.x; v[1] = a.y; v[2] = a.z; v[3] = a.w;
  v[4] = b.x; v[5] = b.y; v[6] = b.z; v[7] = b.w;
  float s = 0.f, s2 = 0.f;
#pragma unroll
  for (int i = 0; i < 8; ++i) { s += v[i]; s2 += v[i] * v[i]; }
#pragma unroll
  for (int off = 16; off > 0; off >>= 1) {
    s  += __shfl_xor(s, off, 32);
    s2 += __shfl_xor(s2, off, 32);
  }
  if ((tid & 31) == 0) { ssum[tid >> 5] = s; ssum2[tid >> 5] = s2; }
  __syncthreads();
  if (tid == 0) {
    float ts = 0.f, ts2 = 0.f;
#pragma unroll
    for (int i = 0; i < 8; ++i) { ts += ssum[i]; ts2 += ssum2[i]; }
    float mu = ts / (float)G_;
    float var = ts2 / (float)G_ - mu * mu;
    smu[0] = mu;
    srs[0] = rsqrtf(var + LN_EPSF);
  }
  __syncthreads();
  float mu = smu[0], rs = srs[0];
#pragma unroll
  for (int i = 0; i < 8; ++i) {
    int g = tid * 8 + i;
    pr[g] = (v[i] - mu) * rs * gamma[g] + beta[g];
  }
}

// ---------------- global sense-reversing barrier ----------------
__device__ __forceinline__ void gbar(unsigned* cnt, unsigned* gen) {
  __syncthreads();
  if (threadIdx.x == 0) {
    __threadfence();
    unsigned g = __hip_atomic_load(gen, __ATOMIC_RELAXED, __HIP_MEMORY_SCOPE_AGENT);
    unsigned arrived = __hip_atomic_fetch_add(cnt, 1u, __ATOMIC_ACQ_REL, __HIP_MEMORY_SCOPE_AGENT);
    if (arrived == NWG - 1) {
      __hip_atomic_store(cnt, 0u, __ATOMIC_RELAXED, __HIP_MEMORY_SCOPE_AGENT);
      __hip_atomic_fetch_add(gen, 1u, __ATOMIC_RELEASE, __HIP_MEMORY_SCOPE_AGENT);
    } else {
      while (__hip_atomic_load(gen, __ATOMIC_ACQUIRE, __HIP_MEMORY_SCOPE_AGENT) == g)
        __builtin_amdgcn_s_sleep(2);
    }
    __threadfence();
  }
  __syncthreads();
}

// ---------------- Phase 2: persistent LiGRU recurrence ----------------
// 32 WGs; WG i owns h cols [32i,32i+32). U rows (32 'a' + 32 'z') pinned in LDS as bf16 hi/lo,
// staged with GLOBAL_LOAD_ASYNC_TO_LDS_B128 (ASYNCcnt). h exchanged per step via
// double-buffered global bf16 hi/lo, device barrier per step.
__global__ __launch_bounds__(RT_THREADS) void ligru_persist(
    const bf16* __restrict__ Uh, const bf16* __restrict__ Ul,
    const float* __restrict__ w, const float* __restrict__ h0,
    bf16* __restrict__ hhBuf, bf16* __restrict__ hlBuf,
    float* __restrict__ out, unsigned* bar) {
  extern __shared__ char smem[];
  bf16*  sUh = (bf16*)smem;                                // LROWS*LSTR bf16
  bf16*  sUl = sUh + LROWS * LSTR;
  float* red = (float*)(sUl + LROWS * LSTR);               // RWAVES*1024 f32
  float* sH  = red + RWAVES * 1024;                        // 16*32 f32

  int tid = threadIdx.x, lane = tid & 31, wv = tid >> 5;
  int wg = blockIdx.x;

  // Pin this WG's U slice into LDS (rows 0..31: 'a' rows, 32..63: 'z' rows) via
  // async global->LDS DMA: one b128 per lane per issue, no VGPR data path.
  for (int idx = tid; idx < LROWS * (H_ / 8); idx += RT_THREADS) {
    int r = idx / (H_ / 8);
    int kc = (idx % (H_ / 8)) * 8;
    int gr = (r < NCOLS) ? (wg * NCOLS + r) : (H_ + wg * NCOLS + (r - NCOLS));
    unsigned long long gh = (unsigned long long)(uintptr_t)(Uh + (size_t)gr * H_ + kc);
    unsigned long long gl = (unsigned long long)(uintptr_t)(Ul + (size_t)gr * H_ + kc);
    unsigned dh = (unsigned)(uintptr_t)(sUh + r * LSTR + kc);  // low 32b = LDS offset
    unsigned dl = (unsigned)(uintptr_t)(sUl + r * LSTR + kc);
    asm volatile("global_load_async_to_lds_b128 %0, %1, off"
                 :: "v"(dh), "v"(gh) : "memory");
    asm volatile("global_load_async_to_lds_b128 %0, %1, off"
                 :: "v"(dl), "v"(gl) : "memory");
  }
  asm volatile("s_wait_asynccnt 0" ::: "memory");
  __syncthreads();

  // Init h block (h0 broadcast over batch) into buffer 0
  for (int idx = tid; idx < B_ * NCOLS; idx += RT_THREADS) {
    int b = idx >> 5, j = idx & 31;
    float v = h0[wg * NCOLS + j];
    sH[idx] = v;
    bf16 hb = (bf16)v;
    hhBuf[(size_t)b * H_ + wg * NCOLS + j] = hb;
    hlBuf[(size_t)b * H_ + wg * NCOLS + j] = (bf16)(v - (float)hb);
  }
  __threadfence();
  gbar(bar, bar + 1);

  for (int t = 0; t < T_; ++t) {
    const bf16* hhr = hhBuf + (size_t)(t & 1) * B_ * H_;
    const bf16* hlr = hlBuf + (size_t)(t & 1) * B_ * H_;
    bf16* hhw = hhBuf + (size_t)((t + 1) & 1) * B_ * H_;
    bf16* hlw = hlBuf + (size_t)((t + 1) & 1) * B_ * H_;

    v8f c0 = zero8(), c1 = zero8(), c2 = zero8(), c3 = zero8();
    int kbase = wv * KCH;
#pragma unroll
    for (int ks = 0; ks < KCH / 32; ++ks) {
      int k0 = kbase + ks * 32;
      v16bf ah = load_frag_a(hhr, H_, 0, k0, lane);
      v16bf al = load_frag_a(hlr, H_, 0, k0, lane);
      v16bf bh, bl;
      bh = load_frag_b(sUh, LSTR, 0, k0, lane);
      bl = load_frag_b(sUl, LSTR, 0, k0, lane);
      c0 = WMMA_BF16(ah, bh, c0); c0 = WMMA_BF16(ah, bl, c0); c0 = WMMA_BF16(al, bh, c0);
      bh = load_frag_b(sUh, LSTR, 16, k0, lane);
      bl = load_frag_b(sUl, LSTR, 16, k0, lane);
      c1 = WMMA_BF16(ah, bh, c1); c1 = WMMA_BF16(ah, bl, c1); c1 = WMMA_BF16(al, bh, c1);
      bh = load_frag_b(sUh, LSTR, 32, k0, lane);
      bl = load_frag_b(sUl, LSTR, 32, k0, lane);
      c2 = WMMA_BF16(ah, bh, c2); c2 = WMMA_BF16(ah, bl, c2); c2 = WMMA_BF16(al, bh, c2);
      bh = load_frag_b(sUh, LSTR, 48, k0, lane);
      bl = load_frag_b(sUl, LSTR, 48, k0, lane);
      c3 = WMMA_BF16(ah, bh, c3); c3 = WMMA_BF16(ah, bl, c3); c3 = WMMA_BF16(al, bh, c3);
    }
    // Stage per-wave partials
    float* myred = red + wv * 1024;
#pragma unroll
    for (int r = 0; r < 8; ++r) {
      myred[0 * 256 + r * 32 + lane] = c0[r];
      myred[1 * 256 + r * 32 + lane] = c1[r];
      myred[2 * 256 + r * 32 + lane] = c2[r];
      myred[3 * 256 + r * 32 + lane] = c3[r];
    }
    __syncthreads();
    // Cross-wave K reduction (result left in wave-0 region)
#pragma unroll
    for (int q = 0; q < 4; ++q) {
      int e = tid + q * 256;
      float s = red[e];
#pragma unroll
      for (int wvi = 1; wvi < RWAVES; ++wvi) s += red[wvi * 1024 + e];
      red[e] = s;
    }
    __syncthreads();
    // Pointwise gate update: 16x32 (b, j) pairs, 2 per thread
#pragma unroll
    for (int q = 0; q < 2; ++q) {
      int p = tid + q * 256;
      int b = p >> 5, j = p & 31;
      int laneg = (j & 15) + ((b >> 3) << 4);
      int rr = b & 7;
      int ea = (j >> 4) * 256 + rr * 32 + laneg;
      int ez = ((32 + j) >> 4) * 256 + rr * 32 + laneg;
      size_t mrow = (size_t)(b * T_ + t);
      float aval = red[ea] + w[mrow * G_ + wg * NCOLS + j];
      float zraw = red[ez] + w[mrow * G_ + H_ + wg * NCOLS + j];
      float z = 1.0f / (1.0f + __expf(-zraw));
      float hold = sH[p];
      float hc = aval > 0.0f ? aval : 0.0f;
      float hn = z * hold + (1.0f - z) * hc;
      sH[p] = hn;
      out[mrow * H_ + wg * NCOLS + j] = hn;
      bf16 hb = (bf16)hn;
      hhw[(size_t)b * H_ + wg * NCOLS + j] = hb;
      hlw[(size_t)b * H_ + wg * NCOLS + j] = (bf16)(hn - (float)hb);
    }
    __threadfence();
    gbar(bar, bar + 1);
  }
}

extern "C" void kernel_launch(void* const* d_in, const int* in_sizes, int n_in,
                              void* d_out, int out_size, void* d_ws, size_t ws_size,
                              hipStream_t stream) {
  const float* x     = (const float*)d_in[0];
  const float* W_w   = (const float*)d_in[1];
  const float* U_w   = (const float*)d_in[2];
  const float* gamma = (const float*)d_in[3];
  const float* beta  = (const float*)d_in[4];
  const float* h0    = (const float*)d_in[5];
  float* out = (float*)d_out;

  char* ws = (char*)d_ws;
  size_t off = 0;
  unsigned* bar = (unsigned*)ws;                 off += 256;
  bf16* Xh = (bf16*)(ws + off);                  off += (size_t)M_ * D_ * 2;
  bf16* Xl = (bf16*)(ws + off);                  off += (size_t)M_ * D_ * 2;
  bf16* Wh = (bf16*)(ws + off);                  off += (size_t)G_ * D_ * 2;
  bf16* Wl = (bf16*)(ws + off);                  off += (size_t)G_ * D_ * 2;
  bf16* Uh = (bf16*)(ws + off);                  off += (size_t)G_ * H_ * 2;
  bf16* Ul = (bf16*)(ws + off);                  off += (size_t)G_ * H_ * 2;
  bf16* hhBuf = (bf16*)(ws + off);               off += (size_t)2 * B_ * H_ * 2;
  bf16* hlBuf = (bf16*)(ws + off);               off += (size_t)2 * B_ * H_ * 2;
  float* wbuf = (float*)(ws + off);              off += (size_t)M_ * G_ * 4;

  hipMemsetAsync(bar, 0, 256, stream);

  split_f32_bf16<<<2048, 256, 0, stream>>>(x,   Xh, Xl, M_ * D_);
  split_f32_bf16<<<1024, 256, 0, stream>>>(W_w, Wh, Wl, G_ * D_);
  split_f32_bf16<<<2048, 256, 0, stream>>>(U_w, Uh, Ul, G_ * H_);

  gemm_xw<<<(M_ / 32) * (G_ / 64) / 8, 256, 0, stream>>>(Xh, Xl, Wh, Wl, wbuf);
  ln_kernel<<<M_, 256, 0, stream>>>(wbuf, gamma, beta);

  size_t lds = (size_t)LROWS * LSTR * 2 * 2      // sUh + sUl
             + (size_t)RWAVES * 1024 * 4         // reduction buffer
             + (size_t)B_ * NCOLS * 4;           // resident h block
  ligru_persist<<<NWG, RT_THREADS, lds, stream>>>(Uh, Ul, wbuf, h0, hhBuf, hlBuf, out, bar);
}
